// ContextualAttention_13434657702216
// MI455X (gfx1250) — matrix operand order repaired and checked by reading, not compiled, not run on hardware
//
#include <hip/hip_runtime.h>
#include <hip/hip_bf16.h>
#include <stddef.h>
#include <stdint.h>

// MI455X / gfx1250: wave32, WMMA 16x16x32 f16->f32, TDM + async global->LDS.
typedef __attribute__((ext_vector_type(16))) _Float16 v16h;
typedef __attribute__((ext_vector_type(8)))  _Float16 v8h;
typedef __attribute__((ext_vector_type(8)))  float    v8f;
typedef __attribute__((ext_vector_type(4))) unsigned int v4u;
typedef __attribute__((ext_vector_type(4))) int v4i;
typedef __attribute__((ext_vector_type(8))) int v8i;

#define BATCH 2
#define HF 128
#define WF 128
#define CH 64
#define HL 64
#define WL 64
#define LQ 4096      // HL*WL
#define KC 576       // 3*3*CH

// ---------------- workspace layout (bytes), total ~91.5 MB ----------------
constexpr size_t OFF_F16F = 0;                               // f16 low-res features
constexpr size_t SZ_F16F  = (size_t)BATCH*HL*WL*CH*2;
constexpr size_t OFF_XF16 = OFF_F16F + SZ_F16F;              // f16 full-res x
constexpr size_t SZ_XF16  = (size_t)BATCH*HF*WF*CH*2;
constexpr size_t OFF_M    = OFF_XF16 + SZ_XF16;              // pooled mask
constexpr size_t SZ_M     = (size_t)BATCH*HL*WL*4;
constexpr size_t OFF_SQ   = OFF_M + SZ_M;                    // per-pixel sum f^2
constexpr size_t SZ_SQ    = SZ_M;
constexpr size_t OFF_INVN = OFF_SQ + SZ_SQ;                  // 1/patch-norm
constexpr size_t SZ_INVN  = (size_t)BATCH*LQ*4;
constexpr size_t OFF_MM   = OFF_INVN + SZ_INVN;              // mask gate
constexpr size_t SZ_MM    = SZ_INVN;
constexpr size_t OFF_YI   = OFF_MM + SZ_MM;                  // f16 softmax attention
constexpr size_t SZ_YI    = (size_t)BATCH*LQ*LQ*2;
constexpr size_t OFF_R    = OFF_YI + SZ_YI;                  // recon GEMM result
constexpr size_t SZ_R     = (size_t)BATCH*LQ*KC*4;

__device__ __forceinline__ int tmap64(int i) {               // 64x64 flat transpose
    return ((i & 63) << 6) | (i >> 6);
}

// Build a v16h A-fragment (contiguous k = 16*half + s, same order as B) from two contiguous runs.
__device__ __forceinline__ v16h frag_cat(v8h lo, v8h hi) {
    return __builtin_shufflevector(lo, hi, 0, 1, 2, 3, 4, 5, 6, 7,
                                           8, 9, 10, 11, 12, 13, 14, 15);
}

// -------- x (f32) -> f16 copy --------
__global__ void k_half(const float* __restrict__ x, _Float16* __restrict__ xh, int n) {
    int i = blockIdx.x * blockDim.x + threadIdx.x;
    if (i < n) xh[i] = (_Float16)x[i];
}

// -------- 2x2 avgpool of x and mask, per-pixel sum of squares --------
__global__ void k_pool(const float* __restrict__ x, const float* __restrict__ mask,
                       _Float16* __restrict__ f16f, float* __restrict__ m,
                       float* __restrict__ sq) {
    int blk = blockIdx.x;                 // b*4096 + yh*64 + yw
    int b  = blk >> 12;
    int yh = (blk >> 6) & 63;
    int yw = blk & 63;
    int c  = threadIdx.x;                 // 64 threads
    size_t base = ((size_t)(b * HF + 2 * yh) * WF + 2 * yw) * CH + c;
    float v = 0.25f * (x[base] + x[base + CH] +
                       x[base + (size_t)WF * CH] + x[base + (size_t)WF * CH + CH]);
    size_t o = ((size_t)(b * HL + yh) * WL + yw) * CH + c;
    f16f[o] = (_Float16)v;
    __shared__ float red[64];
    red[c] = v * v;
    __syncthreads();
    if (c == 0) {
        float s = 0.f;
        for (int i = 0; i < 64; ++i) s += red[i];
        size_t om = (size_t)(b * HL + yh) * WL + yw;
        sq[om] = s;
        size_t mb = (size_t)(b * HF + 2 * yh) * WF + 2 * yw;
        m[om] = 0.25f * (mask[mb] + mask[mb + 1] + mask[mb + WF] + mask[mb + WF + 1]);
    }
}

// -------- per-patch inverse norm + exact-ones mask gate --------
__global__ void k_gates(const float* __restrict__ sq, const float* __restrict__ m,
                        float* __restrict__ invn, float* __restrict__ mmg) {
    int i = blockIdx.x * blockDim.x + threadIdx.x;    // b*4096 + q
    if (i >= BATCH * LQ) return;
    int b = i >> 12, q = i & (LQ - 1);
    int qh = q >> 6, qw = q & 63;
    float ss = 0.f, ms = 0.f;
    for (int dh = -1; dh <= 1; ++dh)
        for (int dw = -1; dw <= 1; ++dw) {
            int y = qh + dh, x2 = qw + dw;
            if ((unsigned)y < (unsigned)HL && (unsigned)x2 < (unsigned)WL) {
                size_t o = (size_t)(b * HL + y) * WL + x2;
                ss += sq[o];
                ms += m[o];
            }
        }
    invn[i] = 1.0f / fmaxf(sqrtf(ss), 1e-4f);
    mmg[i]  = ((ms / 9.0f) == 1.0f) ? 1.0f : 0.0f;    // patch mean exactly 1.0
}

// -------- score GEMM: S = im2col(f) * im2col(f)^T * diag(invn) --------
__global__ void __launch_bounds__(256) k_score(const _Float16* __restrict__ f16f,
                                               const float* __restrict__ invn,
                                               float* __restrict__ att) {
    __shared__ __align__(64) _Float16 aS[64 * 32];   // [row][K] K-contiguous
    __shared__ __align__(64) _Float16 bS[64 * 32];   // [col][K] K-contiguous
    const int b = blockIdx.z;
    const int rowBase = blockIdx.y << 6;
    const int colBase = blockIdx.x << 6;
    const int tid  = threadIdx.x;
    const int lane = tid & 31;
    const int wave = tid >> 5;
    const int t0 = wave * 2;
    const int rg = t0 >> 2, cg0 = t0 & 3, cg1 = cg0 + 1;
    const _Float16* fb = f16f + (size_t)b * HL * WL * CH;

    const int frow = tid >> 2;            // 0..63 tile row (A) / col (B)
    const int fko  = (tid & 3) * 8;       // K sub-offset
    const int pA = rowBase + frow, phA = pA >> 6, pwA = pA & 63;
    const int pB = colBase + frow, phB = pB >> 6, pwB = pB & 63;

    v8f acc0 = {}; v8f acc1 = {};
    for (int kbase = 0; kbase < KC; kbase += 32) {
        const int d  = kbase >> 6;        // constant over the chunk
        const int cb = (kbase & 63) + fko;
        const int dh = d / 3, dw = d - dh * 3;
        __syncthreads();
        {   // A tile
            int yy = phA + dh - 1, xx = pwA + dw - 1;
            v8h va = {};
            if ((unsigned)yy < (unsigned)HL && (unsigned)xx < (unsigned)WL)
                va = *(const v8h*)(fb + ((size_t)yy * WL + xx) * CH + cb);
            *(v8h*)(aS + frow * 32 + fko) = va;
        }
        {   // B tile (B = A^T of the q columns)
            int yy = phB + dh - 1, xx = pwB + dw - 1;
            v8h vb = {};
            if ((unsigned)yy < (unsigned)HL && (unsigned)xx < (unsigned)WL)
                vb = *(const v8h*)(fb + ((size_t)yy * WL + xx) * CH + cb);
            *(v8h*)(bS + frow * 32 + fko) = vb;
        }
        if (kbase + 32 < KC) {            // prefetch next chunk (global_prefetch)
            int kn = kbase + 32;
            int dn = kn >> 6, cn = (kn & 63) + fko;
            int dhn = dn / 3, dwn = dn - dhn * 3;
            int yy = phA + dhn - 1, xx = pwA + dwn - 1;
            if ((unsigned)yy < (unsigned)HL && (unsigned)xx < (unsigned)WL)
                __builtin_prefetch(fb + ((size_t)yy * WL + xx) * CH + cn, 0, 3);
        }
        __syncthreads();
        const int half = lane >> 4;
        const _Float16* ar = aS + ((rg << 4) + (lane & 15)) * 32;
        v16h a0 = frag_cat(*(const v8h*)(ar + 16 * half),
                           *(const v8h*)(ar + 8 + 16 * half));
        v16h b0 = *(const v16h*)(bS + ((cg0 << 4) + (lane & 15)) * 32 + 16 * half);
        v16h b1 = *(const v16h*)(bS + ((cg1 << 4) + (lane & 15)) * 32 + 16 * half);
        acc0 = __builtin_amdgcn_wmma_f32_16x16x32_f16(false, a0, false, b0,
                                                      (short)0, acc0, false, false);
        acc1 = __builtin_amdgcn_wmma_f32_16x16x32_f16(false, a0, false, b1,
                                                      (short)0, acc1, false, false);
    }
    const float* ivb = invn + (size_t)b * LQ;
    float* A = att + (size_t)b * LQ * LQ;
    int Mb = rowBase + (rg << 4) + ((lane >> 4) << 3);
    int q0 = colBase + (cg0 << 4) + (lane & 15);
    int q1 = colBase + (cg1 << 4) + (lane & 15);
    float iv0 = ivb[q0], iv1 = ivb[q1];
    for (int r = 0; r < 8; ++r) {
        int p = Mb + r;
        A[(size_t)p * LQ + q0] = acc0[r] * iv0;
        A[(size_t)p * LQ + q1] = acc1[r] * iv1;
    }
}

// -------- fused diagonal-conv x2 + gated softmax; writes f16 attention --------
__global__ void __launch_bounds__(256) k_fuse_softmax(const float* __restrict__ att,
                                                      const float* __restrict__ mmg,
                                                      _Float16* __restrict__ yi) {
    const int p = blockIdx.x;
    const int b = blockIdx.y;
    const int tid = threadIdx.x;
    const float* S   = att + (size_t)b * LQ * LQ;
    const float* mmb = mmg + (size_t)b * LQ;
    _Float16*    Y   = yi  + ((size_t)b * LQ + p) * LQ;
    const int tp = tmap64(p);
    float logit[16];
    float lmax = -3.4e38f;
    for (int i = 0; i < 16; ++i) {
        int q  = tid + (i << 8);
        int tq = tmap64(q);
        float acc = 0.f;
        #pragma unroll
        for (int d2 = -1; d2 <= 1; ++d2) {
            int ap = tp + d2, bp = tq + d2;             // shift, transposed flat order
            if ((unsigned)ap < (unsigned)LQ && (unsigned)bp < (unsigned)LQ) {
                int a2 = tmap64(ap), b2 = tmap64(bp);
                #pragma unroll
                for (int d1 = -1; d1 <= 1; ++d1) {
                    int a3 = a2 + d1, b3 = b2 + d1;     // shift, original flat order
                    if ((unsigned)a3 < (unsigned)LQ && (unsigned)b3 < (unsigned)LQ)
                        acc += S[(size_t)a3 * LQ + b3];
                }
            }
        }
        float g = mmb[q];
        float l = acc * g * 10.0f;
        logit[i] = l;
        lmax = fmaxf(lmax, l);
    }
    __shared__ float red[256];
    red[tid] = lmax;
    __syncthreads();
    for (int s = 128; s > 0; s >>= 1) {
        if (tid < s) red[tid] = fmaxf(red[tid], red[tid + s]);
        __syncthreads();
    }
    float gmax = red[0];
    __syncthreads();
    float ex[16];
    float lsum = 0.f;
    for (int i = 0; i < 16; ++i) { ex[i] = expf(logit[i] - gmax); lsum += ex[i]; }
    red[tid] = lsum;
    __syncthreads();
    for (int s = 128; s > 0; s >>= 1) {
        if (tid < s) red[tid] += red[tid + s];
        __syncthreads();
    }
    float inv = 1.0f / red[0];
    for (int i = 0; i < 16; ++i) {
        int q = tid + (i << 8);
        Y[q] = (_Float16)(ex[i] * inv * mmb[q]);
    }
}

// -------- reconstruction GEMM: R[p][n] = sum_q yi[p][q] * P[q][n] --------
// Two K-chunks per iteration: chunk k -> TDM tensor_load_to_lds (wave 0,
// TENSORcnt), chunk k+32 -> per-lane async DMA (ASYNCcnt). One barrier covers
// both; 4 WMMAs accumulate both chunks.
__global__ void __launch_bounds__(256) k_recon(const _Float16* __restrict__ yi,
                                               const _Float16* __restrict__ xh,
                                               float* __restrict__ R) {
    __shared__ __align__(64) _Float16 aS0[64 * 32];  // [p-row][K(q)] chunk k
    __shared__ __align__(64) _Float16 aS1[64 * 32];  // [p-row][K(q)] chunk k+32
    __shared__ __align__(64) _Float16 bS0[64 * 32];  // [n-col][K(q)] chunk k
    __shared__ __align__(64) _Float16 bS1[64 * 32];  // [n-col][K(q)] chunk k+32
    const int b = blockIdx.z;
    const int rowBase = blockIdx.x << 6;   // p tile
    const int dd = blockIdx.y;             // n tile == one (dh,dw) slice
    const int colBase = dd << 6;
    const int dh = dd / 3, dw = dd - dh * 3;
    const int tid  = threadIdx.x;
    const int lane = tid & 31;
    const int wave = tid >> 5;
    const int t0 = wave * 2;
    const int rg = t0 >> 2, cg0 = t0 & 3, cg1 = cg0 + 1;
    const _Float16* Yb  = yi + (size_t)b * LQ * LQ;
    const _Float16* YbT = Yb + (size_t)rowBase * LQ;         // tile-row base
    const _Float16* Xb  = xh + (size_t)b * HF * WF * CH;

    // D# group 1 (constant): data_size=2B, tensor 4096x4096, tile 32x64,
    // dim0 stride 4096 elements.  (ISA 8.4 bit packing.)
    const v8i g1 = { 0x00010000, 0x10000000, 0x10000000, (32 << 16),
                     64, 4096, 0, 0 };
    const v4i z4 = {};
#if __has_builtin(__builtin_amdgcn_tensor_load_to_lds) && (__clang_major__ >= 23)
    const v8i z8 = {};
#endif
    const int arow = tid >> 2, akoff = (tid & 3) * 8;        // async fill role
    const uint32_t ldsA1 = (uint32_t)(uintptr_t)(aS1 + arow * 32 + akoff);
#if !__has_builtin(__builtin_amdgcn_tensor_load_to_lds)
    const uint32_t ldsA0 = (uint32_t)(uintptr_t)(aS0 + arow * 32 + akoff);
#endif
    const int bq = tid >> 3, bc0 = (tid & 7) * 8;            // B fill role

    v8f acc0 = {}; v8f acc1 = {};
    for (int k0 = 0; k0 < LQ; k0 += 64) {
        __syncthreads();
#if __has_builtin(__builtin_amdgcn_tensor_load_to_lds)
        if (tid < 32) {   // wave 0 drives the TDM for chunk k0
            uint64_t ga = (uint64_t)(uintptr_t)(YbT + k0);
            v4u g0;
            g0.x = 1u;                                   // count=1 (valid D#)
            g0.y = (uint32_t)(uintptr_t)aS0;             // lds_addr
            g0.z = (uint32_t)ga;                         // global_addr[31:0]
            g0.w = (uint32_t)((ga >> 32) & 0x01FFFFFFu) | (2u << 30);  // type=2
#if __clang_major__ >= 23
            __builtin_amdgcn_tensor_load_to_lds(g0, g1, z4, z4, z8, 0);
#else
            __builtin_amdgcn_tensor_load_to_lds(g0, g1, z4, z4, 0);
#endif
        }
#else
        {   // fallback: per-lane async DMA for chunk k0
            uint32_t voff = (uint32_t)(((arow * LQ) + k0 + akoff) * 2);
            asm volatile("global_load_async_to_lds_b128 %0, %1, %2 offset:0"
                         :: "v"(ldsA0), "v"(voff), "s"(YbT) : "memory");
        }
#endif
        {   // chunk k0+32 via per-lane async DMA (ASYNCcnt path)
            uint32_t voff = (uint32_t)(((arow * LQ) + k0 + 32 + akoff) * 2);
            asm volatile("global_load_async_to_lds_b128 %0, %1, %2 offset:0"
                         :: "v"(ldsA1), "v"(voff), "s"(YbT) : "memory");
        }
        #pragma unroll
        for (int h = 0; h < 2; ++h) {   // B tiles: vector load + transposed b16 stores
            int q  = k0 + 32 * h + bq;
            int yy = 2 * (q >> 6) + dh;
            int xx = 2 * (q & 63) + dw;
            v8h vb = {};
            if (yy < HF && xx < WF)
                vb = *(const v8h*)(Xb + ((size_t)yy * WF + xx) * CH + bc0);
            _Float16* bSd = h ? bS1 : bS0;
            #pragma unroll
            for (int i = 0; i < 8; ++i)
                bSd[(bc0 + i) * 32 + bq] = vb[i];
        }
        asm volatile("s_wait_asynccnt 0x0" ::: "memory");
#if __has_builtin(__builtin_amdgcn_tensor_load_to_lds)
        if (tid < 32) __builtin_amdgcn_s_wait_tensorcnt((short)0);
#endif
        __syncthreads();
        const int half = lane >> 4;
        const int aoff = ((rg << 4) + (lane & 15)) * 32;
        const int boff0 = ((cg0 << 4) + (lane & 15)) * 32 + 16 * half;
        const int boff1 = ((cg1 << 4) + (lane & 15)) * 32 + 16 * half;
        {   // chunk k0
            v16h a = frag_cat(*(const v8h*)(aS0 + aoff + 16 * half),
                              *(const v8h*)(aS0 + aoff + 8 + 16 * half));
            v16h b0 = *(const v16h*)(bS0 + boff0);
            v16h b1 = *(const v16h*)(bS0 + boff1);
            acc0 = __builtin_amdgcn_wmma_f32_16x16x32_f16(false, a, false, b0,
                                                          (short)0, acc0, false, false);
            acc1 = __builtin_amdgcn_wmma_f32_16x16x32_f16(false, a, false, b1,
                                                          (short)0, acc1, false, false);
        }
        {   // chunk k0+32
            v16h a = frag_cat(*(const v8h*)(aS1 + aoff + 16 * half),
                              *(const v8h*)(aS1 + aoff + 8 + 16 * half));
            v16h b0 = *(const v16h*)(bS1 + boff0);
            v16h b1 = *(const v16h*)(bS1 + boff1);
            acc0 = __builtin_amdgcn_wmma_f32_16x16x32_f16(false, a, false, b0,
                                                          (short)0, acc0, false, false);
            acc1 = __builtin_amdgcn_wmma_f32_16x16x32_f16(false, a, false, b1,
                                                          (short)0, acc1, false, false);
        }
    }
    float* Rb = R + (size_t)b * LQ * KC;
    int Mb = rowBase + (rg << 4) + ((lane >> 4) << 3);
    int n0 = colBase + (cg0 << 4) + (lane & 15);
    int n1 = colBase + (cg1 << 4) + (lane & 15);
    for (int r = 0; r < 8; ++r) {
        int p = Mb + r;
        Rb[(size_t)p * KC + n0] = acc0[r];
        Rb[(size_t)p * KC + n1] = acc1[r];
    }
}

// -------- transpose-conv scatter: gather <=4 R entries per output pixel --------
__global__ void k_scatter(const float* __restrict__ R, float* __restrict__ y) {
    int idx = blockIdx.x * blockDim.x + threadIdx.x;
    if (idx >= BATCH * HF * WF * CH) return;
    int c  = idx & 63;
    int xw = (idx >> 6) & 127;
    int yy = (idx >> 13) & 127;
    int b  = idx >> 20;
    const float* Rb = R + (size_t)b * LQ * KC;
    float acc = 0.f;
    for (int dh = 0; dh < 3; ++dh) {
        int t = yy - dh;
        if (t < 0 || (t & 1)) continue;
        int ph = t >> 1;
        if (ph >= HL) continue;
        for (int dw = 0; dw < 3; ++dw) {
            int u = xw - dw;
            if (u < 0 || (u & 1)) continue;
            int pw = u >> 1;
            if (pw >= WL) continue;
            acc += Rb[(size_t)((ph << 6) + pw) * KC + (dh * 3 + dw) * 64 + c];
        }
    }
    y[idx] = acc * 0.25f;
}

extern "C" void kernel_launch(void* const* d_in, const int* in_sizes, int n_in,
                              void* d_out, int out_size, void* d_ws, size_t ws_size,
                              hipStream_t stream) {
    const float* x    = (const float*)d_in[0];   // [2,128,128,64]
    const float* mask = (const float*)d_in[1];   // [2,128,128,1]
    float* out     = (float*)d_out;
    float* out_y   = out;                                   // [2,128,128,64]
    float* out_att = out + (size_t)BATCH * HF * WF * CH;    // [2,64,64,4096]

    char* ws = (char*)d_ws;                      // requires ws_size >= ~91.5 MB
    _Float16* f16f = (_Float16*)(ws + OFF_F16F);
    _Float16* xh   = (_Float16*)(ws + OFF_XF16);
    float*    m    = (float*)   (ws + OFF_M);
    float*    sq   = (float*)   (ws + OFF_SQ);
    float*    invn = (float*)   (ws + OFF_INVN);
    float*    mmg  = (float*)   (ws + OFF_MM);
    _Float16* yi   = (_Float16*)(ws + OFF_YI);
    float*    R    = (float*)   (ws + OFF_R);

    const int nX = BATCH * HF * WF * CH;
    k_half<<<(nX + 255) / 256, 256, 0, stream>>>(x, xh, nX);
    k_pool<<<BATCH * HL * WL, 64, 0, stream>>>(x, mask, f16f, m, sq);
    k_gates<<<(BATCH * LQ + 255) / 256, 256, 0, stream>>>(sq, m, invn, mmg);
    k_score<<<dim3(LQ / 64, LQ / 64, BATCH), 256, 0, stream>>>(f16f, invn, out_att);
    k_fuse_softmax<<<dim3(LQ, BATCH), 256, 0, stream>>>(out_att, mmg, yi);
    k_recon<<<dim3(LQ / 64, KC / 64, BATCH), 256, 0, stream>>>(yi, xh, R);
    k_scatter<<<(nX + 255) / 256, 256, 0, stream>>>(R, out_y);
}
